// Encoder_44676249813590
// MI455X (gfx1250) — matrix operand order repaired and checked by reading, not compiled, not run on hardware
//
#include <hip/hip_runtime.h>

// ---------------------------------------------------------------------------
// Types for CDNA5 WMMA (wave32): bf16 A/B operands, f32 accumulator.
// ---------------------------------------------------------------------------
typedef __bf16 v4bf  __attribute__((ext_vector_type(4)));
typedef __bf16 v8bf  __attribute__((ext_vector_type(8)));
typedef __bf16 v16bf __attribute__((ext_vector_type(16)));
typedef float  v8f   __attribute__((ext_vector_type(8)));

typedef int v4i __attribute__((vector_size(16)));                 // 16-byte chunk
typedef __attribute__((address_space(1))) v4i gv4i;               // global AS
typedef __attribute__((address_space(3))) v4i lv4i;               // LDS AS

constexpr int cL = 4;
constexpr int cB = 32;
constexpr int cT = 256;
constexpr int cD = 512;
constexpr int cU = 1024;
constexpr int cG = 4096;          // 4*U
constexpr int cM = cB * cT;       // 8192 rows for the big GEMMs

__device__ __forceinline__ float sigf(float x)     { return 1.f / (1.f + __expf(-x)); }
__device__ __forceinline__ float tanhfastf(float x){ return 1.f - 2.f / (__expf(2.f * x) + 1.f); }

__device__ __forceinline__ v16bf cat16(v8bf lo, v8bf hi) {
  return __builtin_shufflevector(lo, hi, 0,1,2,3,4,5,6,7,8,9,10,11,12,13,14,15);
}

__device__ __forceinline__ v8f wmma_bf16(v16bf a, v16bf b, v8f c) {
  return __builtin_amdgcn_wmma_f32_16x16x32_bf16(false, a, false, b, (short)0, c, false, false);
}

// ---------------------------------------------------------------------------
// Async global->LDS copy (CDNA5 GLOBAL_LOAD_ASYNC_TO_LDS_B128, ASYNCcnt) with
// a compile-safe synchronous fallback if the builtin is absent.
// ---------------------------------------------------------------------------
#if __has_builtin(__builtin_amdgcn_global_load_async_to_lds_b128)
#define HAS_ASYNC_LDS 1
#else
#define HAS_ASYNC_LDS 0
#endif

__device__ __forceinline__ void cp16_g2l(void* lds, const void* gptr) {
#if HAS_ASYNC_LDS
  __builtin_amdgcn_global_load_async_to_lds_b128(
      (gv4i*)(void*)gptr, (lv4i*)lds, 0, 0);
#else
  *(v8bf*)lds = *(const v8bf*)gptr;
#endif
}

__device__ __forceinline__ void cp_wait() {
#if HAS_ASYNC_LDS
#if __has_builtin(__builtin_amdgcn_s_wait_asynccnt)
  __builtin_amdgcn_s_wait_asynccnt(0);
#else
  asm volatile("s_wait_asynccnt 0x0" ::: "memory");
#endif
#endif
}

// ---------------------------------------------------------------------------
// fp32 -> bf16 flat convert (vectorized x4)
// ---------------------------------------------------------------------------
__global__ void k_cvt(const float* __restrict__ s, __bf16* __restrict__ d, int n4) {
  int i = blockIdx.x * blockDim.x + threadIdx.x;
  if (i < n4) {
    float4 f = ((const float4*)s)[i];
    v4bf o;
    o[0] = (__bf16)f.x; o[1] = (__bf16)f.y; o[2] = (__bf16)f.z; o[3] = (__bf16)f.w;
    ((v4bf*)d)[i] = o;
  }
}

// ---------------------------------------------------------------------------
// fp32 (K,N) row-major -> bf16 (N,K) row-major (transposed), LDS-tiled.
// ---------------------------------------------------------------------------
__global__ void k_tcvt(const float* __restrict__ src, __bf16* __restrict__ dst, int K, int N) {
  __shared__ float tile[32][33];
  int n0 = blockIdx.x * 32, k0 = blockIdx.y * 32;
  int tx = threadIdx.x & 31, ty = threadIdx.x >> 5;
#pragma unroll
  for (int i = 0; i < 32; i += 8)
    tile[ty + i][tx] = src[(size_t)(k0 + ty + i) * N + n0 + tx];
  __syncthreads();
#pragma unroll
  for (int i = 0; i < 32; i += 8)
    dst[(size_t)(n0 + ty + i) * K + k0 + tx] = (__bf16)tile[tx][ty + i];
}

// ---------------------------------------------------------------------------
// Reset h ping-pong buffers and the grid-barrier counter.
// ---------------------------------------------------------------------------
__global__ void k_reset(__bf16* __restrict__ h, unsigned* __restrict__ bar, int n) {
  int i = blockIdx.x * blockDim.x + threadIdx.x;
  if (i == 0) *bar = 0u;
  for (int j = i; j < n; j += gridDim.x * blockDim.x) h[j] = (__bf16)0.f;
}

// ---------------------------------------------------------------------------
// Tiled bf16 WMMA GEMM:  C[M,N](f32,+bias) = A[M,K](bf16) @ Bt[N,K](bf16)^T
// 256 threads = 8 waves (2 M-rows x 4 N-cols); wave tile 32x64 (2x4 WMMA).
// A tile double-buffered in LDS; staging overlapped with compute (async path
// if available); one workgroup barrier per K-chunk.
// ---------------------------------------------------------------------------
__global__ __launch_bounds__(256)
void k_gemm(const __bf16* __restrict__ A, const __bf16* __restrict__ Bt,
            const float* __restrict__ bias, float* __restrict__ Cf,
            __bf16* __restrict__ Cb, int M, int N, int K) {
  __shared__ __bf16 atile[2][64 * 32];
  const int tid  = threadIdx.x;
  const int wave = tid >> 5, lane = tid & 31;
  const int lo = lane & 15, hi = lane >> 4;
  const int wm = wave >> 2, wn = wave & 3;
  const int mBlk = blockIdx.y * 64, nBlk = blockIdx.x * 256;

  const v8f vzero = {0.f, 0.f, 0.f, 0.f, 0.f, 0.f, 0.f, 0.f};
  v8f acc[2][4];
#pragma unroll
  for (int i = 0; i < 2; ++i)
#pragma unroll
    for (int j = 0; j < 4; ++j) acc[i][j] = vzero;

  const int r = tid >> 2, seg = tid & 3;  // 64 rows x 4 segments of 8 bf16
  const __bf16* aRow = A + (size_t)(mBlk + r) * K + seg * 8;
  __bf16* sSlot0 = &atile[0][r * 32 + seg * 8];
  __bf16* sSlot1 = &atile[1][r * 32 + seg * 8];

  const int nk = K >> 5;
  cp16_g2l(sSlot0, aRow);                 // stage chunk 0
  cp_wait();
  __syncthreads();

  for (int kc = 0; kc < nk; ++kc) {
    const int cur = kc & 1;
    if (kc + 1 < nk)                      // stage chunk kc+1 into other buffer
      cp16_g2l(cur ? sSlot0 : sSlot1, aRow + (kc + 1) * 32);

    v16bf a[2];
#pragma unroll
    for (int mt = 0; mt < 2; ++mt) {
      const __bf16* p = &atile[cur][(wm * 32 + mt * 16 + lo) * 32];
      a[mt] = cat16(*(const v8bf*)(p + hi * 8), *(const v8bf*)(p + 16 + hi * 8));
    }
#pragma unroll
    for (int nt = 0; nt < 4; ++nt) {
      int n = nBlk + wn * 64 + nt * 16 + lo;
      const __bf16* q = &Bt[(size_t)n * K + kc * 32];
      v16bf bv = cat16(*(const v8bf*)(q + hi * 8), *(const v8bf*)(q + 16 + hi * 8));
      acc[0][nt] = wmma_bf16(a[0], bv, acc[0][nt]);
      acc[1][nt] = wmma_bf16(a[1], bv, acc[1][nt]);
    }
    cp_wait();
    __syncthreads();
  }

#pragma unroll
  for (int mt = 0; mt < 2; ++mt)
#pragma unroll
    for (int nt = 0; nt < 4; ++nt)
#pragma unroll
      for (int j = 0; j < 8; ++j) {
        int m = mBlk + wm * 32 + mt * 16 + hi * 8 + j;
        int n = nBlk + wn * 64 + nt * 16 + lo;
        float v = acc[mt][nt][j] + (bias ? bias[n] : 0.f);
        Cf[(size_t)m * N + n] = v;
        if (Cb) Cb[(size_t)m * N + n] = (__bf16)v;
      }
}

// ---------------------------------------------------------------------------
// Persistent per-layer LSTM scan. Grid = 8 WGs x 256 threads (64 waves).
// Wave owns 16 units for ALL 4 gates (gating wave-local, c-state in regs).
// h ping-pongs in global bf16; one grid barrier/step; h staged to LDS with
// the fill latency hidden behind the xg accumulator preload.
// ---------------------------------------------------------------------------
__global__ __launch_bounds__(256)
void k_scan(const __bf16* __restrict__ wrt,     // [4U][U] bf16 (N-major)
            const float*  __restrict__ xg,      // [(b*T+t)*4U + n] f32
            float*        __restrict__ seq_f,   // [(b*T+t)*U + u] f32
            __bf16*       __restrict__ seq_b,   // bf16 copy for next GEMM
            __bf16*       __restrict__ hbuf,    // [2][B][U] bf16 ping-pong
            float*        __restrict__ out_h,   // [B][U]
            float*        __restrict__ out_c,   // [B][U]
            unsigned*     __restrict__ bar) {
  __shared__ __bf16 hl[cB * cU];                // 64 KB
  const int tid  = threadIdx.x;
  const int wave = tid >> 5, lane = tid & 31;
  const int lo = lane & 15, hi = lane >> 4;
  const int grp = blockIdx.x * 8 + wave;        // 0..63 unit-groups
  const int u0  = grp * 16;
  const unsigned nGrid = gridDim.x;

  float cst[2][8];
#pragma unroll
  for (int mt = 0; mt < 2; ++mt)
#pragma unroll
    for (int j = 0; j < 8; ++j) cst[mt][j] = 0.f;

  const v8f vzero = {0.f, 0.f, 0.f, 0.f, 0.f, 0.f, 0.f, 0.f};

#pragma unroll 1
  for (int t = 0; t < cT; ++t) {
    // ---- issue h(t-1) -> LDS staging first (latency hidden by xg loads) ----
    const __bf16* hsrc = hbuf + (size_t)(t & 1) * cB * cU;
#pragma unroll
    for (int i = 0; i < 16; ++i)
      cp16_g2l(&hl[(size_t)(tid + 256 * i) * 8], hsrc + (size_t)(tid + 256 * i) * 8);

    // ---- accumulators preloaded with xg[:, t, gate*U + u0 + lo] ----
    v8f acc[4][2];
#pragma unroll
    for (int g = 0; g < 4; ++g)
#pragma unroll
      for (int mt = 0; mt < 2; ++mt) {
        acc[g][mt] = vzero;
#pragma unroll
        for (int j = 0; j < 8; ++j) {
          int m = mt * 16 + hi * 8 + j;
          acc[g][mt][j] = xg[((size_t)m * cT + t) * cG + g * cU + u0 + lo];
        }
      }

    cp_wait();
    __syncthreads();

    // ---- K loop: g += h @ Wr (8 WMMAs per 32-wide K chunk) ----
#pragma unroll 2
    for (int kc = 0; kc < cU / 32; ++kc) {
      v16bf a[2];
#pragma unroll
      for (int mt = 0; mt < 2; ++mt) {
        const __bf16* p = &hl[(mt * 16 + lo) * cU + kc * 32];
        a[mt] = cat16(*(const v8bf*)(p + hi * 8), *(const v8bf*)(p + 16 + hi * 8));
      }
#pragma unroll
      for (int g = 0; g < 4; ++g) {
        int n = g * cU + u0 + lo;
        const __bf16* q = &wrt[(size_t)n * cU + kc * 32];
        v16bf bv = cat16(*(const v8bf*)(q + hi * 8), *(const v8bf*)(q + 16 + hi * 8));
        acc[g][0] = wmma_bf16(a[0], bv, acc[g][0]);
        acc[g][1] = wmma_bf16(a[1], bv, acc[g][1]);
      }
    }

    // ---- gates, state update, h/residual writes ----
    __bf16* hw = hbuf + (size_t)((t + 1) & 1) * cB * cU;
#pragma unroll
    for (int mt = 0; mt < 2; ++mt)
#pragma unroll
      for (int j = 0; j < 8; ++j) {
        int m = mt * 16 + hi * 8 + j;
        int u = u0 + lo;
        float ig = sigf(acc[0][mt][j]);
        float fg = sigf(acc[1][mt][j]);
        float cn = tanhfastf(acc[2][mt][j]);
        float og = sigf(acc[3][mt][j]);
        float cnew = fg * cst[mt][j] + ig * cn;
        cst[mt][j] = cnew;
        float hnew = og * tanhfastf(cnew);
        hw[(size_t)m * cU + u] = (__bf16)hnew;
        size_t sidx = ((size_t)m * cT + t) * cU + u;
        float s = seq_f[sidx] + hnew;
        seq_f[sidx] = s;
        seq_b[sidx] = (__bf16)s;
        if (t == cT - 1) {
          out_h[(size_t)m * cU + u] = hnew;
          out_c[(size_t)m * cU + u] = cnew;
        }
      }

    // ---- grid-wide barrier (release/acquire, agent scope) ----
    __builtin_amdgcn_fence(__ATOMIC_RELEASE, "agent");
    __syncthreads();
    if (tid == 0) {
      __hip_atomic_fetch_add(bar, 1u, __ATOMIC_RELEASE, __HIP_MEMORY_SCOPE_AGENT);
      unsigned target = nGrid * (unsigned)(t + 1);
      while (__hip_atomic_load(bar, __ATOMIC_ACQUIRE, __HIP_MEMORY_SCOPE_AGENT) < target) {
        __builtin_amdgcn_s_sleep(2);
      }
    }
    __syncthreads();
    __builtin_amdgcn_fence(__ATOMIC_ACQUIRE, "agent");
  }
}

// ---------------------------------------------------------------------------
// Host-side orchestration.
// ---------------------------------------------------------------------------
extern "C" void kernel_launch(void* const* d_in, const int* in_sizes, int n_in,
                              void* d_out, int out_size, void* d_ws, size_t ws_size,
                              hipStream_t stream) {
  const float* x     = (const float*)d_in[0];   // (B,T,D)
  const float* Wskip = (const float*)d_in[1];   // (D,U)
  const float* Wk    = (const float*)d_in[2];   // (L,U,4U)
  const float* Wr    = (const float*)d_in[3];   // (L,U,4U)
  const float* bias  = (const float*)d_in[4];   // (L,4U)
  float* out = (float*)d_out;                   // (2,L,B,U)

  char* ws = (char*)d_ws;
  size_t off = 0;
  auto alloc = [&](size_t bytes) -> char* {
    char* p = ws + off;
    off = (off + bytes + 255) & ~(size_t)255;
    return p;
  };

  __bf16* xbf     = (__bf16*)alloc((size_t)cM * cD * 2);
  __bf16* wskip_t = (__bf16*)alloc((size_t)cU * cD * 2);
  __bf16* wk_t    = (__bf16*)alloc((size_t)cL * cG * cU * 2);
  __bf16* wr_t    = (__bf16*)alloc((size_t)cL * cG * cU * 2);
  float*  seq_f   = (float*) alloc((size_t)cM * cU * 4);
  __bf16* seq_b   = (__bf16*)alloc((size_t)cM * cU * 2);
  float*  xg      = (float*) alloc((size_t)cM * cG * 4);
  __bf16* hbuf    = (__bf16*)alloc((size_t)2 * cB * cU * 2);
  unsigned* bar   = (unsigned*)alloc(256);
  (void)ws_size; (void)in_sizes; (void)n_in; (void)out_size;

  {
    int n4 = cM * cD / 4;
    k_cvt<<<(n4 + 255) / 256, 256, 0, stream>>>(x, xbf, n4);
  }
  k_tcvt<<<dim3(cU / 32, cD / 32), 256, 0, stream>>>(Wskip, wskip_t, cD, cU);
  for (int l = 0; l < cL; ++l) {
    k_tcvt<<<dim3(cG / 32, cU / 32), 256, 0, stream>>>(
        Wk + (size_t)l * cU * cG, wk_t + (size_t)l * cG * cU, cU, cG);
    k_tcvt<<<dim3(cG / 32, cU / 32), 256, 0, stream>>>(
        Wr + (size_t)l * cU * cG, wr_t + (size_t)l * cG * cU, cU, cG);
  }
  // Skip projection: seq = x @ W_skip (f32 + bf16 copies).
  k_gemm<<<dim3(cU / 256, cM / 64), 256, 0, stream>>>(
      xbf, wskip_t, nullptr, seq_f, seq_b, cM, cU, cD);
  // Layers.
  for (int l = 0; l < cL; ++l) {
    k_gemm<<<dim3(cG / 256, cM / 64), 256, 0, stream>>>(
        seq_b, wk_t + (size_t)l * cG * cU, bias + (size_t)l * cG, xg, nullptr,
        cM, cG, cU);
    k_reset<<<64, 256, 0, stream>>>(hbuf, bar, 2 * cB * cU);
    k_scan<<<8, 256, 0, stream>>>(
        wr_t + (size_t)l * cG * cU, xg, seq_f, seq_b, hbuf,
        out + (size_t)l * cB * cU, out + (size_t)(cL + l) * cB * cU, bar);
  }
}
